// EdgeAttention_43061342109998
// MI455X (gfx1250) — matrix operand order repaired and checked by reading, not compiled, not run on hardware
//
#include <hip/hip_runtime.h>
#include <hip/hip_bf16.h>

typedef __attribute__((ext_vector_type(16))) __bf16 v16bf;
typedef __attribute__((ext_vector_type(8)))  __bf16 v8bf;
typedef __attribute__((ext_vector_type(8)))  float  v8f;

#define B_ 32
#define N_ 512
#define D_ 128
#define T_ 8
#define NEG_SLOPE_ 0.2f
#define LDU (D_ + 8)   // padded LDS row stride (bf16 units): 272B rows -> conflict-free 16B chunks

// ------------------------------------------------------------------
// Prep: f32 -> bf16 for src/dst; transpose a[d,e,t] -> a_bf[t][e][d]
// ------------------------------------------------------------------
__global__ __launch_bounds__(256)
void EdgeAttn_prep_kernel(const float* __restrict__ src,
                          const float* __restrict__ dst,
                          const float* __restrict__ a,
                          __bf16* __restrict__ src_bf,
                          __bf16* __restrict__ dst_bf,
                          __bf16* __restrict__ a_bf) {
  const int idx0   = blockIdx.x * blockDim.x + threadIdx.x;
  const int stride = gridDim.x * blockDim.x;

  const int n_sd = B_ * N_ * D_;
  for (int i = idx0; i < n_sd; i += stride) {
    src_bf[i] = (__bf16)src[i];
    dst_bf[i] = (__bf16)dst[i];
  }
  const int n_a = D_ * D_ * T_;
  for (int o = idx0; o < n_a; o += stride) {
    const int d = o & (D_ - 1);
    const int e = (o >> 7) & (D_ - 1);
    const int t = o >> 14;
    a_bf[o] = (__bf16)a[(d * D_ + e) * T_ + t];
  }
}

// ------------------------------------------------------------------
// Fragment loaders, per CDNA5 ISA 7.12.2 16-bit layouts (wave32)
// ------------------------------------------------------------------
__device__ __forceinline__ v16bf make16(v8bf lo, v8bf hi) {
  v16bf r;
#pragma unroll
  for (int x = 0; x < 8; ++x) { r[x] = lo[x]; r[x + 8] = hi[x]; }
  return r;
}

// A-matrix 16x32 (MxK): lane holds row m = lane%16.
// lanes 0-15: K = {k0..k0+7, k0+16..k0+23}; lanes 16-31: +8 on both chunks.
// row_base points at &A[m][k0] for this lane (16B aligned).
__device__ __forceinline__ v16bf load_a_frag(const __bf16* row_base, int lane) {
  const int o = (lane & 16) ? 8 : 0;
  v8bf lo = *(const v8bf*)(row_base + o);
  v8bf hi = *(const v8bf*)(row_base + 16 + o);
  return make16(lo, hi);
}

// B-matrix 32x16 (KxN): lane holds column n = lane%16:
// lanes 0-15: K = {k0..k0+7, k0+16..k0+23}; lanes 16-31: +8 on both chunks.
// base points at &B_colmajor[n][k0 + 8*half] for this lane (16B aligned).
__device__ __forceinline__ v16bf load_b_frag(const __bf16* base) {
  v8bf lo = *(const v8bf*)(base);
  v8bf hi = *(const v8bf*)(base + 16);
  return make16(lo, hi);
}

// ------------------------------------------------------------------
// Main fused kernel: per (b, 16-row i-tile) workgroup.
// Phase 1: wave t computes U_t = src_tile(16x128) x a_t(128x128) -> LDS (bf16)
// Phase 2: each wave: 4 j-tiles; 8 t x 4 k-step WMMAs; gather/mask/leaky.
// ------------------------------------------------------------------
__global__ __launch_bounds__(256)
void EdgeAttn_main_kernel(const __bf16* __restrict__ src_bf,
                          const __bf16* __restrict__ dst_bf,
                          const __bf16* __restrict__ a_bf,
                          const int*    __restrict__ edges,
                          float*        __restrict__ out) {
  __shared__ __bf16 U[T_][16][LDU];

  const int wg   = blockIdx.x;          // b*32 + i_tile
  const int b    = wg >> 5;
  const int i0   = (wg & 31) * 16;
  const int lane = threadIdx.x & 31;
  const int wave = threadIdx.x >> 5;    // 0..7
  const int hh   = (lane >> 4);         // 0 = lanes 0-15, 1 = lanes 16-31
  const int l16  = lane & 15;

  // ---------------- Phase 1: U[wave] = src_tile x a_wave ----------------
  {
    const int t = wave;
    const __bf16* srcrow = src_bf + ((size_t)b * N_ + (i0 + l16)) * D_;
    v16bf afrag[4];
#pragma unroll
    for (int kc = 0; kc < 4; ++kc)
      afrag[kc] = load_a_frag(srcrow + kc * 32, lane);

    const __bf16* at = a_bf + (size_t)t * D_ * D_;   // [e][d], d contiguous
#pragma unroll
    for (int nt = 0; nt < 8; ++nt) {
      const int e = nt * 16 + l16;                   // this lane's output column
      v8f acc = {};
#pragma unroll
      for (int kc = 0; kc < 4; ++kc) {
        v16bf bfrag = load_b_frag(at + (size_t)e * D_ + kc * 32 + hh * 8);
        acc = __builtin_amdgcn_wmma_f32_16x16x32_bf16(false, afrag[kc], false, bfrag,
                                                      (short)0, acc, false, false);
      }
      // C layout: reg r -> (m = r + 8*hh, n = l16); store bf16 into LDS
#pragma unroll
      for (int r = 0; r < 8; ++r)
        U[t][r + hh * 8][nt * 16 + l16] = (__bf16)acc[r];
    }
  }
  __syncthreads();

  // ---------------- Phase 2: score tiles, gather by edge type ----------------
  for (int q = 0; q < 4; ++q) {
    const int j0 = (wave * 4 + q) * 16;

    // B-frags of dst^T: lane's column j = j0 + l16, contiguous d
    const __bf16* dstrow = dst_bf + ((size_t)b * N_ + (j0 + l16)) * D_;
    v16bf bfrag[4];
#pragma unroll
    for (int kc = 0; kc < 4; ++kc)
      bfrag[kc] = load_b_frag(dstrow + kc * 32 + hh * 8);

    v8f acc[T_];
#pragma unroll
    for (int t = 0; t < T_; ++t) {
      v8f c = {};
#pragma unroll
      for (int kc = 0; kc < 4; ++kc) {
        const __bf16* arow = &U[t][l16][kc * 32 + ((lane & 16) ? 8 : 0)];
        v8bf lo = *(const v8bf*)(arow);
        v8bf hi = *(const v8bf*)(arow + 16);
        v16bf af = make16(lo, hi);
        c = __builtin_amdgcn_wmma_f32_16x16x32_bf16(false, af, false, bfrag[kc],
                                                    (short)0, c, false, false);
      }
      acc[t] = c;
    }

    // gather selected type, mask -1 -> -inf, leaky relu, store
#pragma unroll
    for (int r = 0; r < 8; ++r) {
      const int i = i0 + r + hh * 8;
      const int j = j0 + l16;
      const int eraw = edges[((size_t)b * N_ + i) * N_ + j];
      const int e = (eraw < 0) ? 0 : eraw;
      const float v0 = acc[0][r], v1 = acc[1][r], v2 = acc[2][r], v3 = acc[3][r];
      const float v4 = acc[4][r], v5 = acc[5][r], v6 = acc[6][r], v7 = acc[7][r];
      const float m0 = (e & 1) ? v1 : v0;
      const float m1 = (e & 1) ? v3 : v2;
      const float m2 = (e & 1) ? v5 : v4;
      const float m3 = (e & 1) ? v7 : v6;
      const float n0 = (e & 2) ? m1 : m0;
      const float n1 = (e & 2) ? m3 : m2;
      float s       = (e & 4) ? n1 : n0;
      s = (eraw < 0) ? -__builtin_huge_valf() : s;
      s = (s >= 0.0f) ? s : NEG_SLOPE_ * s;
      out[((size_t)b * N_ + i) * N_ + j] = s;
    }
  }
}

// ------------------------------------------------------------------
extern "C" void kernel_launch(void* const* d_in, const int* in_sizes, int n_in,
                              void* d_out, int out_size, void* d_ws, size_t ws_size,
                              hipStream_t stream) {
  const float* src   = (const float*)d_in[0];
  const float* dst   = (const float*)d_in[1];
  const float* a     = (const float*)d_in[2];
  const int*   edges = (const int*)d_in[3];
  float* out = (float*)d_out;

  const size_t SB = (size_t)B_ * N_ * D_;           // 2,097,152 elems
  __bf16* src_bf = (__bf16*)d_ws;
  __bf16* dst_bf = src_bf + SB;
  __bf16* a_bf   = dst_bf + SB;                     // total ~8.6 MB of ws

  EdgeAttn_prep_kernel<<<1024, 256, 0, stream>>>(src, dst, a, src_bf, dst_bf, a_bf);
  EdgeAttn_main_kernel<<<B_ * (N_ / 16), 256, 0, stream>>>(src_bf, dst_bf, a_bf, edges, out);
}
